// WindowedViTGen_14439680049540
// MI455X (gfx1250) — compile-verified, matrix operand order checked by reading, not run on hardware
//
#include <hip/hip_runtime.h>
#include <hip/hip_bf16.h>
#include <stddef.h>

// ---------------------------------------------------------------------------
// Windowed ViT generator forward on gfx1250 (MI455X).
// GEMMs run on v_wmma_f32_16x16x32_bf16 with 2x2 register blocking per wave;
// attention (<=9 neighbors) on VALU.
// ---------------------------------------------------------------------------

typedef __attribute__((ext_vector_type(16))) __bf16 v16bf;
typedef __attribute__((ext_vector_type(8)))  __bf16 v8bf;
typedef __attribute__((ext_vector_type(8)))  float  v8f;

#define VIT_N     1024          // patches
#define VIT_D     512
#define VIT_H3    1536          // 3*D
#define VIT_FF    1024
#define VIT_HEADS 8
#define VIT_HD    64
#define VIT_DEPTH 6
#define GRID_W    32            // 32x32 patch grid

// ---------------------------------------------------------------------------
// fp32 -> bf16 conversion (weights, activations)
// ---------------------------------------------------------------------------
__global__ __launch_bounds__(256) void cvt_f32_bf16(const float* __restrict__ in,
                                                    __bf16* __restrict__ out, int n) {
  int i = blockIdx.x * 256 + threadIdx.x;
  if (i < n) out[i] = (__bf16)in[i];
}

// ---------------------------------------------------------------------------
// LayerNorm over D=512, one wave per row, bf16 output (feeds WMMA GEMM).
// ---------------------------------------------------------------------------
__global__ __launch_bounds__(256) void ln_rows_bf16(const float* __restrict__ x,
                                                    const float* __restrict__ gamma,
                                                    const float* __restrict__ beta,
                                                    __bf16* __restrict__ out) {
  const int lane = threadIdx.x & 31;
  const int wid  = threadIdx.x >> 5;
  const int row  = blockIdx.x * 8 + wid;        // 128 blocks x 8 waves = 1024 rows
  const float* xr = x + (size_t)row * VIT_D;

  float vals[16];
  float s = 0.f;
#pragma unroll
  for (int i = 0; i < 16; ++i) { vals[i] = xr[lane + 32 * i]; s += vals[i]; }
#pragma unroll
  for (int m = 16; m >= 1; m >>= 1) s += __shfl_xor(s, m, 32);
  const float mean = s * (1.0f / VIT_D);

  float v = 0.f;
#pragma unroll
  for (int i = 0; i < 16; ++i) { float d = vals[i] - mean; v += d * d; }
#pragma unroll
  for (int m = 16; m >= 1; m >>= 1) v += __shfl_xor(v, m, 32);
  const float rstd = rsqrtf(v * (1.0f / VIT_D) + 1e-5f);

  __bf16* orow = out + (size_t)row * VIT_D;
#pragma unroll
  for (int i = 0; i < 16; ++i) {
    const int d = lane + 32 * i;
    orow[d] = (__bf16)((vals[i] - mean) * rstd * gamma[d] + beta[d]);
  }
}

// ---------------------------------------------------------------------------
// Fragment load: two contiguous b128 runs per the CDNA5 16x16x32 bf16 layout.
//   lane L: K-half = (L>>4)*8; elems 0..7 = K k0+kh..+7, elems 8..15 = +16.
// ---------------------------------------------------------------------------
__device__ __forceinline__ v16bf load_frag(const __bf16* p) {
  const v8bf lo = *(const v8bf*)(p);
  const v8bf hi = *(const v8bf*)(p + 16);
  v16bf f;
#pragma unroll
  for (int i = 0; i < 8; ++i) { f[i] = lo[i]; f[8 + i] = hi[i]; }
  return f;
}

// ---------------------------------------------------------------------------
// Epilogue for one 16x16 C tile: bias + optional residual/ReLU, fp32/bf16 out.
//   C/D layout: lane L, vgpr r -> row (L>>4)*8 + r, col (L&15).
// ---------------------------------------------------------------------------
template <bool RELU, bool RES, bool WF32, bool WBF16>
__device__ __forceinline__ void store_tile(const v8f& acc, int orow0, int ocol,
                                           const float* __restrict__ bias,
                                           const float* __restrict__ res,
                                           float* __restrict__ outF,
                                           __bf16* __restrict__ outB, int M) {
  const float bv = bias[ocol];
#pragma unroll
  for (int r = 0; r < 8; ++r) {
    const size_t o = (size_t)(orow0 + r) * M + ocol;
    float v = acc[r] + bv;
    if (RES)  v += res[o];
    if (RELU) v = v > 0.f ? v : 0.f;
    if (WF32)  outF[o] = v;
    if (WBF16) outB[o] = (__bf16)v;
  }
}

// ---------------------------------------------------------------------------
// WMMA GEMM:  C[rows=1024, M] = A[rows, K](bf16) x W[M, K]^T(bf16) + bias
// 2x2 register blocking: each wave owns a 32x32 C tile (4 accumulators),
// so each A/B fragment feeds two WMMAs -> 4 wmma per 8 b128 loads.
// Block = 256 threads = 8 waves = 128 rows x 64 cols. grid = (M/64, rows/128).
// ---------------------------------------------------------------------------
template <bool RELU, bool RES, bool WF32, bool WBF16>
__global__ __launch_bounds__(256) void gemm_bf16_wmma(
    const __bf16* __restrict__ A,     // [1024, K]
    const __bf16* __restrict__ W,     // [M, K]
    const float*  __restrict__ bias,  // [M]
    const float*  __restrict__ res,   // [1024, M] or unused
    float*        __restrict__ outF,  // [1024, M] or unused
    __bf16*       __restrict__ outB,  // [1024, M] or unused
    int K, int M) {
  const int lane  = threadIdx.x & 31;
  const int wid   = threadIdx.x >> 5;
  const int wr    = wid & 3;          // row tile inside block (0..3) -> 32 rows each
  const int wc    = wid >> 2;         // col tile inside block (0..1) -> 32 cols each
  const int row0  = blockIdx.y * 128 + wr * 32;
  const int col0  = blockIdx.x * 64 + wc * 32;
  const int nloc  = lane & 15;
  const int khalf = lane >> 4;

  const __bf16* pa0 = A + (size_t)(row0 + nloc) * K + khalf * 8;
  const __bf16* pa1 = pa0 + (size_t)16 * K;
  const __bf16* pw0 = W + (size_t)(col0 + nloc) * K + khalf * 8;
  const __bf16* pw1 = pw0 + (size_t)16 * K;

  v8f c00 = {}, c01 = {}, c10 = {}, c11 = {};
  for (int k0 = 0; k0 < K; k0 += 32) {
    if (k0 + 32 < K) {                       // gfx1250 global_prefetch_b8
      __builtin_prefetch(pa0 + k0 + 32, 0, 1);
      __builtin_prefetch(pa1 + k0 + 32, 0, 1);
      __builtin_prefetch(pw0 + k0 + 32, 0, 1);
      __builtin_prefetch(pw1 + k0 + 32, 0, 1);
    }
    const v16bf a0 = load_frag(pa0 + k0);
    const v16bf a1 = load_frag(pa1 + k0);
    const v16bf b0 = load_frag(pw0 + k0);
    const v16bf b1 = load_frag(pw1 + k0);
    c00 = __builtin_amdgcn_wmma_f32_16x16x32_bf16(false, a0, false, b0, (short)0, c00, false, false);
    c01 = __builtin_amdgcn_wmma_f32_16x16x32_bf16(false, a0, false, b1, (short)0, c01, false, false);
    c10 = __builtin_amdgcn_wmma_f32_16x16x32_bf16(false, a1, false, b0, (short)0, c10, false, false);
    c11 = __builtin_amdgcn_wmma_f32_16x16x32_bf16(false, a1, false, b1, (short)0, c11, false, false);
  }

  const int rbase0 = row0 + khalf * 8;        // first tile row block
  const int rbase1 = rbase0 + 16;
  const int cbase0 = col0 + nloc;
  const int cbase1 = cbase0 + 16;
  store_tile<RELU, RES, WF32, WBF16>(c00, rbase0, cbase0, bias, res, outF, outB, M);
  store_tile<RELU, RES, WF32, WBF16>(c01, rbase0, cbase1, bias, res, outF, outB, M);
  store_tile<RELU, RES, WF32, WBF16>(c10, rbase1, cbase0, bias, res, outF, outB, M);
  store_tile<RELU, RES, WF32, WBF16>(c11, rbase1, cbase1, bias, res, outF, outB, M);
}

// ---------------------------------------------------------------------------
// Windowed attention (WIN=1 -> <=9 neighbors), one wave per (patch, head).
// Lane handles 2 head-dim elements. Scores via shfl reductions, register
// softmax, fused PV accumulate. Output bf16 [N, D] (feeds out-proj GEMM).
// qkv layout: [N, 1536] = [q | k | v] per row, head h at offset h*64.
// ---------------------------------------------------------------------------
__global__ __launch_bounds__(256) void win_attn_bf16(const float* __restrict__ qkv,
                                                     __bf16* __restrict__ out) {
  const int lane  = threadIdx.x & 31;
  const int wid   = threadIdx.x >> 5;
  const int g     = blockIdx.x * 8 + wid;      // 0 .. N*HEADS-1
  const int head  = g & (VIT_HEADS - 1);
  const int patch = g >> 3;
  const int pr = patch >> 5, pc = patch & 31;

  const float2 q2 = *(const float2*)(qkv + (size_t)patch * VIT_H3 + head * VIT_HD + lane * 2);

  float sc[9];
#pragma unroll
  for (int di = -1; di <= 1; ++di) {
#pragma unroll
    for (int dj = -1; dj <= 1; ++dj) {
      const int  idx   = (di + 1) * 3 + (dj + 1);
      const int  nr    = pr + di, nc = pc + dj;
      const bool valid = ((unsigned)nr < (unsigned)GRID_W) &&
                         ((unsigned)nc < (unsigned)GRID_W);
      float p = 0.f;
      if (valid) {
        const float2 k2 = *(const float2*)(qkv + (size_t)(nr * GRID_W + nc) * VIT_H3 +
                                           VIT_D + head * VIT_HD + lane * 2);
        p = q2.x * k2.x + q2.y * k2.y;
      }
#pragma unroll
      for (int m = 16; m >= 1; m >>= 1) p += __shfl_xor(p, m, 32);
      sc[idx] = valid ? p * 0.125f : -3.0e38f;   // 1/sqrt(64) scale; -inf mask
    }
  }

  float mx = sc[0];
#pragma unroll
  for (int i = 1; i < 9; ++i) mx = fmaxf(mx, sc[i]);
  float e[9], sum = 0.f;
#pragma unroll
  for (int i = 0; i < 9; ++i) { e[i] = __expf(sc[i] - mx); sum += e[i]; }
  const float inv = 1.0f / sum;

  float2 o = make_float2(0.f, 0.f);
#pragma unroll
  for (int di = -1; di <= 1; ++di) {
#pragma unroll
    for (int dj = -1; dj <= 1; ++dj) {
      const int  idx   = (di + 1) * 3 + (dj + 1);
      const int  nr    = pr + di, nc = pc + dj;
      const bool valid = ((unsigned)nr < (unsigned)GRID_W) &&
                         ((unsigned)nc < (unsigned)GRID_W);
      if (valid) {
        const float  p  = e[idx] * inv;
        const float2 v2 = *(const float2*)(qkv + (size_t)(nr * GRID_W + nc) * VIT_H3 +
                                           2 * VIT_D + head * VIT_HD + lane * 2);
        o.x += p * v2.x; o.y += p * v2.y;
      }
    }
  }

  __bf16* op = out + (size_t)patch * VIT_D + head * VIT_HD + lane * 2;
  op[0] = (__bf16)o.x;
  op[1] = (__bf16)o.y;
}

// ---------------------------------------------------------------------------
// Pixel head: y[1024,192] -> sigmoid -> out[3,256,256]
// out[c, h*8+pi, w*8+pj] = sigmoid(y[h*32+w, c*64 + pi*8 + pj])
// ---------------------------------------------------------------------------
__global__ __launch_bounds__(256) void pixel_sigmoid(const float* __restrict__ y,
                                                     float* __restrict__ out) {
  const int idx = blockIdx.x * 256 + threadIdx.x;   // 0 .. 3*256*256-1
  const int c   = idx >> 16;
  const int rem = idx & 0xFFFF;
  const int yi  = rem >> 8;
  const int xi  = rem & 255;
  const int h = yi >> 3, pi = yi & 7;
  const int w = xi >> 3, pj = xi & 7;
  const int patch = h * GRID_W + w;
  const int feat  = c * 64 + pi * 8 + pj;
  const float v = y[(size_t)patch * 192 + feat];
  out[idx] = 1.0f / (1.0f + __expf(-v));
}

// ---------------------------------------------------------------------------
// Host-side orchestration
// ---------------------------------------------------------------------------
static inline size_t alignup(size_t v) { return (v + 255) & ~(size_t)255; }

extern "C" void kernel_launch(void* const* d_in, const int* in_sizes, int n_in,
                              void* d_out, int out_size, void* d_ws, size_t ws_size,
                              hipStream_t stream) {
  (void)in_sizes; (void)n_in; (void)out_size; (void)ws_size;

  const float* pos    = (const float*)d_in[0];   // [1,1024,512]
  const float* ln1_g  = (const float*)d_in[1];   // [6,512]
  const float* ln1_b  = (const float*)d_in[2];
  const float* qkv_w  = (const float*)d_in[3];   // [6,1536,512]
  const float* qkv_b  = (const float*)d_in[4];   // [6,1536]
  const float* out_w  = (const float*)d_in[5];   // [6,512,512]
  const float* out_b  = (const float*)d_in[6];
  const float* ln2_g  = (const float*)d_in[7];
  const float* ln2_b  = (const float*)d_in[8];
  const float* ffn_w1 = (const float*)d_in[9];   // [6,1024,512]
  const float* ffn_b1 = (const float*)d_in[10];  // [6,1024]
  const float* ffn_w2 = (const float*)d_in[11];  // [6,512,1024]
  const float* ffn_b2 = (const float*)d_in[12];
  const float* pix_w  = (const float*)d_in[13];  // [192,512]
  const float* pix_b  = (const float*)d_in[14];  // [192]

  // ---- workspace layout ----
  char*  ws  = (char*)d_ws;
  size_t off = 0;
  float*  x      = (float*)(ws + off);  off = alignup(off + (size_t)VIT_N * VIT_D * 4);
  __bf16* h_bf   = (__bf16*)(ws + off); off = alignup(off + (size_t)VIT_N * VIT_D * 2);
  float*  qkvbuf = (float*)(ws + off);  off = alignup(off + (size_t)VIT_N * VIT_H3 * 4);
  __bf16* att_bf = (__bf16*)(ws + off); off = alignup(off + (size_t)VIT_N * VIT_D * 2);
  __bf16* hf_bf  = (__bf16*)(ws + off); off = alignup(off + (size_t)VIT_N * VIT_FF * 2);
  float*  ybuf   = (float*)(ws + off);  off = alignup(off + (size_t)VIT_N * 192 * 4);
  __bf16* qkvw_b = (__bf16*)(ws + off); off = alignup(off + (size_t)VIT_DEPTH * VIT_H3 * VIT_D * 2);
  __bf16* outw_b = (__bf16*)(ws + off); off = alignup(off + (size_t)VIT_DEPTH * VIT_D * VIT_D * 2);
  __bf16* fw1_b  = (__bf16*)(ws + off); off = alignup(off + (size_t)VIT_DEPTH * VIT_FF * VIT_D * 2);
  __bf16* fw2_b  = (__bf16*)(ws + off); off = alignup(off + (size_t)VIT_DEPTH * VIT_D * VIT_FF * 2);
  __bf16* pixw_b = (__bf16*)(ws + off); off = alignup(off + (size_t)192 * VIT_D * 2);

  // ---- weight down-conversion to bf16 (once per call) ----
  {
    auto cvt = [&](const float* s, __bf16* d, int n) {
      cvt_f32_bf16<<<(n + 255) / 256, 256, 0, stream>>>(s, d, n);
    };
    cvt(qkv_w,  qkvw_b, VIT_DEPTH * VIT_H3 * VIT_D);
    cvt(out_w,  outw_b, VIT_DEPTH * VIT_D * VIT_D);
    cvt(ffn_w1, fw1_b,  VIT_DEPTH * VIT_FF * VIT_D);
    cvt(ffn_w2, fw2_b,  VIT_DEPTH * VIT_D * VIT_FF);
    cvt(pix_w,  pixw_b, 192 * VIT_D);
  }

  // x = pos_embed
  hipMemcpyAsync(x, pos, (size_t)VIT_N * VIT_D * sizeof(float),
                 hipMemcpyDeviceToDevice, stream);

  const dim3 blk(256);
  const dim3 gLN(VIT_N / 8);                 // 128 blocks, 8 rows each
  const dim3 gAtt(VIT_N * VIT_HEADS / 8);    // 1024 blocks, 8 (patch,head) each

  for (int l = 0; l < VIT_DEPTH; ++l) {
    const __bf16* Wqkv = qkvw_b + (size_t)l * VIT_H3 * VIT_D;
    const __bf16* Wout = outw_b + (size_t)l * VIT_D * VIT_D;
    const __bf16* Wf1  = fw1_b  + (size_t)l * VIT_FF * VIT_D;
    const __bf16* Wf2  = fw2_b  + (size_t)l * VIT_D * VIT_FF;

    // h1 = LN1(x) -> bf16
    ln_rows_bf16<<<gLN, blk, 0, stream>>>(x, ln1_g + l * VIT_D, ln1_b + l * VIT_D, h_bf);

    // qkv = h1 @ qkv_w^T + qkv_b          [1024, 1536] fp32
    gemm_bf16_wmma<false, false, true, false>
        <<<dim3(VIT_H3 / 64, VIT_N / 128), blk, 0, stream>>>(
            h_bf, Wqkv, qkv_b + (size_t)l * VIT_H3, nullptr, qkvbuf, nullptr,
            VIT_D, VIT_H3);

    // windowed attention -> bf16 [1024, 512]
    win_attn_bf16<<<gAtt, blk, 0, stream>>>(qkvbuf, att_bf);

    // x = x + attn @ out_w^T + out_b      (residual fused)
    gemm_bf16_wmma<false, true, true, false>
        <<<dim3(VIT_D / 64, VIT_N / 128), blk, 0, stream>>>(
            att_bf, Wout, out_b + (size_t)l * VIT_D, x, x, nullptr,
            VIT_D, VIT_D);

    // h2 = LN2(x) -> bf16
    ln_rows_bf16<<<gLN, blk, 0, stream>>>(x, ln2_g + l * VIT_D, ln2_b + l * VIT_D, h_bf);

    // hf = relu(h2 @ ffn_w1^T + b1) -> bf16 [1024, 1024]
    gemm_bf16_wmma<true, false, false, true>
        <<<dim3(VIT_FF / 64, VIT_N / 128), blk, 0, stream>>>(
            h_bf, Wf1, ffn_b1 + (size_t)l * VIT_FF, nullptr, nullptr, hf_bf,
            VIT_D, VIT_FF);

    // x = x + hf @ ffn_w2^T + b2          (residual fused)
    gemm_bf16_wmma<false, true, true, false>
        <<<dim3(VIT_D / 64, VIT_N / 128), blk, 0, stream>>>(
            hf_bf, Wf2, ffn_b2 + (size_t)l * VIT_D, x, x, nullptr,
            VIT_FF, VIT_D);
  }

  // pixel head: y = x @ pix_w^T + pix_b   [1024, 192]
  cvt_f32_bf16<<<(VIT_N * VIT_D + 255) / 256, 256, 0, stream>>>(x, h_bf, VIT_N * VIT_D);
  gemm_bf16_wmma<false, false, true, false>
      <<<dim3(192 / 64, VIT_N / 128), blk, 0, stream>>>(
          h_bf, pixw_b, pix_b, nullptr, ybuf, nullptr, VIT_D, 192);

  // un-patchify + sigmoid -> [3, 256, 256]
  pixel_sigmoid<<<(3 * 256 * 256) / 256, blk, 0, stream>>>(ybuf, (float*)d_out);
}
